// LSTM_69818988364525
// MI455X (gfx1250) — matrix operand order repaired and checked by reading, not compiled, not run on hardware
//
#include <hip/hip_runtime.h>

// Problem dims
#define TT     512
#define BB     64
#define INPUT  512
#define HH     1024
#define KDIM   (HH + INPUT)     // 1536
#define KPAD   (KDIM + 8)       // pad LDS rows (16B-aligned rows, de-conflict banks)
#define NWG    128
#define JS     8                // hidden columns owned per workgroup
#define COLS   32               // 4 gates * JS output columns per workgroup
#define THREADS 256             // 8 wave32
#define KH     32               // k-steps covering h   (K 0..1023)
#define KX     16               // k-steps covering x_t (K 1024..1535)

typedef __bf16 bf16;
typedef __attribute__((ext_vector_type(8)))  __bf16 v8bf;
typedef __attribute__((ext_vector_type(16))) __bf16 v16bf;
typedef __attribute__((ext_vector_type(8)))  float  v8f;

__device__ __forceinline__ bf16 f2bf(float f) {
  unsigned u = __builtin_bit_cast(unsigned, f);
  unsigned r = (u + 0x7FFFu + ((u >> 16) & 1u)) >> 16;   // round-to-nearest-even
  unsigned short s = (unsigned short)r;
  return __builtin_bit_cast(bf16, s);
}

// ---------------- prep kernels (run every launch: graph-replay deterministic) --------------

__global__ void init_state(unsigned* counter, bf16* hbuf) {
  int tid = blockIdx.x * blockDim.x + threadIdx.x;
  if (tid == 0) *counter = 0u;
  for (int i = tid; i < BB * HH; i += gridDim.x * blockDim.x) hbuf[i] = f2bf(0.f);
}

__global__ void conv_x(const float* __restrict__ x, bf16* __restrict__ xbf) {
  const int n = TT * BB * INPUT;
  for (int i = blockIdx.x * blockDim.x + threadIdx.x; i < n; i += gridDim.x * blockDim.x)
    xbf[i] = f2bf(x[i]);
}

// Rearrange weights: dst[(wg*COLS + gate*JS + jj)*KDIM + k] = W_gate[wg*JS + jj][k]
__global__ void prep_weights(const float* __restrict__ Wf, const float* __restrict__ Wi,
                             const float* __restrict__ Wc, const float* __restrict__ Wo,
                             const float* __restrict__ bfv, const float* __restrict__ biv,
                             const float* __restrict__ bcv, const float* __restrict__ bov,
                             bf16* __restrict__ wbf, float* __restrict__ bias) {
  const float* W[4] = {Wf, Wi, Wc, Wo};
  const float* bv[4] = {bfv, biv, bcv, bov};
  const int total = 4 * HH * KDIM;
  for (int idx = blockIdx.x * blockDim.x + threadIdx.x; idx < total;
       idx += gridDim.x * blockDim.x) {
    int g   = idx / (HH * KDIM);
    int rem = idx - g * (HH * KDIM);
    int j   = rem / KDIM;
    int k   = rem - j * KDIM;
    int wg  = j / JS, jj = j - wg * JS;
    wbf[(size_t)(wg * COLS + g * JS + jj) * KDIM + k] = f2bf(W[g][rem]);
  }
  for (int idx = blockIdx.x * blockDim.x + threadIdx.x; idx < 4 * HH;
       idx += gridDim.x * blockDim.x) {
    int g = idx / HH, j = idx - g * HH;
    bias[(j / JS) * COLS + g * JS + (j % JS)] = bv[g][j];
  }
}

// ---------------- persistent LSTM kernel --------------------------------------------------

__launch_bounds__(THREADS)
__global__ void lstm_persistent(const bf16* __restrict__ xbf,
                                const bf16* __restrict__ wbf,
                                const float* __restrict__ bias,
                                bf16* __restrict__ hbuf,
                                float* __restrict__ out,
                                unsigned* __restrict__ counter) {
  extern __shared__ char smem[];
  bf16*  sW    = (bf16*)smem;                                         // COLS * KPAD bf16
  float* sGate = (float*)(smem + (size_t)COLS * KPAD * sizeof(bf16)); // BB * COLS f32
  float* sC    = sGate + BB * COLS;                                   // BB * JS f32
  float* sBias = sC + BB * JS;                                        // COLS f32

  const int tid   = threadIdx.x;
  const int wg    = blockIdx.x;
  const int lane  = tid & 31;
  const int wave  = tid >> 5;        // 0..7
  const int mtile = wave & 3;        // batch tile (4 x 16 rows)
  const int ntile = wave >> 2;       // column tile (2 x 16 cols)

  // One-time async DMA of this WG's weight slice into LDS (ASYNCcnt-tracked,
  // VGPR-bypassing global->LDS path). 16 bytes per lane per issue.
  {
    const unsigned ldsbase = (unsigned)(uintptr_t)sW;
    for (int idx = tid; idx < COLS * (KDIM / 8); idx += THREADS) {
      int col = idx / (KDIM / 8);
      int kc  = (idx - col * (KDIM / 8)) * 8;
      unsigned lds = ldsbase + (unsigned)((col * KPAD + kc) * sizeof(bf16));
      unsigned long long ga =
          (unsigned long long)(uintptr_t)(wbf + (size_t)(wg * COLS + col) * KDIM + kc);
      asm volatile("global_load_async_to_lds_b128 %0, %1, off"
                   :: "v"(lds), "v"(ga) : "memory");
    }
    asm volatile("s_wait_asynccnt 0x0" ::: "memory");
  }
  for (int i = tid; i < BB * JS; i += THREADS) sC[i] = 0.f;   // c0 = 0
  if (tid < COLS) sBias[tid] = bias[wg * COLS + tid];
  __syncthreads();

  // Per-lane fragment geometry (contiguous k = 16*(lane/16) + e in both A and B, wave32):
  const int arow = mtile * 16 + (lane & 15);  // A: row = M (batch)
  const int hi16 = lane >> 4;                 // lane group selects K sub-range
  const int col  = ntile * 16 + (lane & 15);  // B/C/D: column = N

  // Constant per-lane base pointers: all per-iteration offsets become immediates.
  const bf16* hbase = hbuf + (size_t)arow * HH + hi16 * 16;      // A, h region
  const bf16* bbase = sW + (size_t)col * KPAD + hi16 * 16;       // B (LDS)
  const float bcol  = sBias[col];
  const int   gate  = col >> 3;               // 0=f, 1=i, 2=c_hat, 3=o

  for (int t = 0; t < TT; ++t) {
    const bf16* xbase = xbf + ((size_t)t * BB + arow) * INPUT + hi16 * 16;

    // Two interleaved accumulator chains (halves the WMMA dependency chain,
    // gives the scheduler room to overlap loads with XDL work).
    v8f acc0 = {}, acc1 = {};

#pragma unroll
    for (int kk = 0; kk < KH; ++kk) {      // K 0..1023 : A from h
      v8bf  a0 = *(const v8bf*)(hbase + kk * 32);
      v8bf  a1 = *(const v8bf*)(hbase + kk * 32 + 8);
      v16bf b  = *(const v16bf*)(bbase + kk * 32);
      v16bf a;
#pragma unroll
      for (int e = 0; e < 8; ++e) { a[e] = a0[e]; a[8 + e] = a1[e]; }
      if (kk & 1)
        acc1 = __builtin_amdgcn_wmma_f32_16x16x32_bf16(false, a, false, b,
                                                       (short)0, acc1, false, false);
      else
        acc0 = __builtin_amdgcn_wmma_f32_16x16x32_bf16(false, a, false, b,
                                                       (short)0, acc0, false, false);
    }
#pragma unroll
    for (int kk = 0; kk < KX; ++kk) {      // K 1024..1535 : A from x_t
      v8bf  a0 = *(const v8bf*)(xbase + kk * 32);
      v8bf  a1 = *(const v8bf*)(xbase + kk * 32 + 8);
      v16bf b  = *(const v16bf*)(bbase + (KH + kk) * 32);
      v16bf a;
#pragma unroll
      for (int e = 0; e < 8; ++e) { a[e] = a0[e]; a[8 + e] = a1[e]; }
      if (kk & 1)
        acc1 = __builtin_amdgcn_wmma_f32_16x16x32_bf16(false, a, false, b,
                                                       (short)0, acc1, false, false);
      else
        acc0 = __builtin_amdgcn_wmma_f32_16x16x32_bf16(false, a, false, b,
                                                       (short)0, acc0, false, false);
    }

    // Bias + activation; C/D layout: VGPR r -> M = r + 8*(lane>=16), N = lane%16.
#pragma unroll
    for (int r = 0; r < 8; ++r) {
      int brow = mtile * 16 + r + hi16 * 8;
      float v = acc0[r] + acc1[r] + bcol;
      float act = (gate == 2) ? tanhf(v) : (1.f / (1.f + __expf(-v)));
      sGate[brow * COLS + col] = act;
    }
    __syncthreads();

    // Fused cell update for this WG's 8 hidden units.
    for (int e = tid; e < BB * JS; e += THREADS) {
      int b_ = e >> 3;
      int jj = e & 7;
      float f  = sGate[b_ * COLS + jj];
      float i  = sGate[b_ * COLS + 8 + jj];
      float cb = sGate[b_ * COLS + 16 + jj];
      float o  = sGate[b_ * COLS + 24 + jj];
      float c  = f * sC[e] + i * cb;
      sC[e] = c;
      float h = o * tanhf(c);
      int jglob = wg * JS + jj;
      hbuf[b_ * HH + jglob] = f2bf(h);                       // bf16 for next-step A
      out[((size_t)t * BB + b_) * HH + jglob] = h;           // f32 result
    }

    if (t + 1 < TT)  // pull next timestep's x slab toward L2 (global_prefetch_b8)
      __builtin_prefetch(xbf + (size_t)(t + 1) * BB * INPUT + tid * 128, 0, 1);

    // Device-wide barrier: all WGs must publish h before anyone starts step t+1.
    __threadfence();
    __syncthreads();
    if (tid == 0) {
      __hip_atomic_fetch_add(counter, 1u, __ATOMIC_RELEASE, __HIP_MEMORY_SCOPE_AGENT);
      const unsigned target = (unsigned)(t + 1) * NWG;
      while (__hip_atomic_load(counter, __ATOMIC_ACQUIRE, __HIP_MEMORY_SCOPE_AGENT) < target)
        __builtin_amdgcn_s_sleep(2);
    }
    __syncthreads();
  }
}

// ---------------- launch ------------------------------------------------------------------

extern "C" void kernel_launch(void* const* d_in, const int* in_sizes, int n_in,
                              void* d_out, int out_size, void* d_ws, size_t ws_size,
                              hipStream_t stream) {
  const float* x   = (const float*)d_in[0];
  const float* Wf  = (const float*)d_in[1];
  const float* Wi  = (const float*)d_in[2];
  const float* Wc  = (const float*)d_in[3];
  const float* Wo  = (const float*)d_in[4];
  const float* bfv = (const float*)d_in[5];
  const float* biv = (const float*)d_in[6];
  const float* bcv = (const float*)d_in[7];
  const float* bov = (const float*)d_in[8];
  float* out = (float*)d_out;

  char* ws = (char*)d_ws;
  unsigned* counter = (unsigned*)ws;                              // 256 B slot
  bf16* hbuf = (bf16*)(ws + 256);                                 // 64*1024*2   = 128 KB
  bf16* xbf  = (bf16*)(ws + 256 + (size_t)BB * HH * 2);           // T*B*IN*2    = 32 MB
  bf16* wbf  = (bf16*)((char*)xbf + (size_t)TT * BB * INPUT * 2); // 4*H*K*2     = 12.6 MB
  float* bias = (float*)((char*)wbf + (size_t)4 * HH * KDIM * 2); // 16 KB

  init_state<<<64, 256, 0, stream>>>(counter, hbuf);
  conv_x<<<2048, 256, 0, stream>>>(x, xbf);
  prep_weights<<<4096, 256, 0, stream>>>(Wf, Wi, Wc, Wo, bfv, biv, bcv, bov, wbf, bias);

  const size_t smem = (size_t)COLS * KPAD * sizeof(bf16) +
                      (size_t)(BB * COLS + BB * JS + COLS) * sizeof(float);
  lstm_persistent<<<NWG, THREADS, smem, stream>>>(xbf, wbf, bias, hbuf, out, counter);
}
